// CMDFuse_6889127543269
// MI455X (gfx1250) — matrix operand order repaired and checked by reading, not compiled, not run on hardware
//
#include <hip/hip_runtime.h>
#include <hip/hip_bf16.h>

// ---------------- problem constants ----------------
static constexpr int S_ = 4;
static constexpr int N_ = 200000;
static constexpr int NV_ = 100000;
static constexpr int M_ = 80000;
static constexpr int C_ = 20;

// ---------------- types ----------------
typedef _Float16 h16;
typedef __attribute__((ext_vector_type(8)))  _Float16 v8h;
typedef __attribute__((ext_vector_type(16))) _Float16 v16h;
typedef __attribute__((ext_vector_type(8)))  float    v8f;

// ---------------- f16 transposed weight offsets in global workspace (halves)
enum : int {
  OFF_LW    = 0,        // leaner_W     [s][o64][i64]
  OFF_LW2   = 16384,    // leaner2_W
  OFF_F1    = 32768,    // fcs1_W       [s][o64][i128]
  OFF_F12   = 65536,    // fcs1_2_W
  OFF_F2A   = 98304,    // fcs2_W rows 0..63   [s][o64][i64]
  OFF_F2A2  = 114688,   // fcs2_2_W rows 0..63
  OFF_H3W1  = 131072,   // head3d W1    [s][o128][i64]
  OFF_H3W2  = 163840,   // head3d W2    [s][o32(pad)][i128]
  OFF_HFW1  = 180224,   // head_fuse W1
  OFF_HFW2  = 212992,   // head_fuse W2
  OFF_HF2W1 = 229376,   // head_fuse2 W1
  OFF_HF2W2 = 262144,   // head_fuse2 W2
  OFF_CLS1  = 278528,   // cls_W1       [o128][i256]
  OFF_CLSW2 = 311296,   // cls_W2       [o32][i128]
  OFF_CLS21 = 315392,   // cls2_W1
  OFF_CLS22 = 348160,   // cls2_W2
  WTOT      = 352256
};

// ---------------- LDS weight-staging offsets for k_main (halves) ----------
// padded strides: K=64 -> 72, K=128 -> 136, K=256 -> 264 (conflict-free)
enum : int {
  LS_LW    = 0,       // 64 x 72   = 4608
  LS_LW2   = 4608,
  LS_F1    = 9216,    // 64 x 136  = 8704
  LS_F12   = 17920,
  LS_F2A   = 26624,   // 64 x 72
  LS_F2A2  = 31232,
  LS_HFW1  = 35840,   // 128 x 72  = 9216
  LS_HF2W1 = 45056,
  LS_HFW2  = 54272,   // 32 x 136  = 4352
  LS_HF2W2 = 58624,
  WS_HALVES = 62976,  // 125952 B  (cls phase reuses [0, 38144))
  LS_CLSW2 = 33792    // cls_W1 at 0 (128 x 264 = 33792), W2 here (32 x 136)
};

// per-wave working LDS layout for k_main (halves)
static constexpr int ST_X   = 72;
static constexpr int ST_CAT = 264;
static constexpr int ST_TB  = 136;
static constexpr int WV_XP   = 0;                   // 16 x 72 = 1152
static constexpr int WV_XG   = 1152;
static constexpr int WV_T1   = 2304;
static constexpr int WV_CAT1 = 3456;                // 16 x 264 = 4224
static constexpr int WV_CAT2 = 7680;
static constexpr int WV_HALVES = 11904;             // 23808 B / wave
// head/cls hidden buffer (16 x 136 = 2176) aliases WV_XP/WV_XG (dead by then)

// ---------------- WMMA helpers ----------------
__device__ __forceinline__ v16h cat8(v8h a, v8h b) {
  return __builtin_shufflevector(a, b, 0,1,2,3,4,5,6,7,8,9,10,11,12,13,14,15);
}

__device__ __forceinline__ v8f wmma_(v16h a, v16h b, v8f c) {
  return __builtin_amdgcn_wmma_f32_16x16x32_f16(false, a, false, b, (short)0, c,
                                                false, false);
}

// A-fragment (16x32 f16) from row-major LDS tile (contiguous k = 16h + s, matches B).
__device__ __forceinline__ v16h load_a_lds(const h16* base, int stride, int kt) {
  int lane = threadIdx.x & 31;
  const h16* p = base + (lane & 15) * stride + kt + 16 * (lane >> 4);
  v8h lo = *(const v8h*)p;
  v8h hi = *(const v8h*)(p + 8);
  return cat8(lo, hi);
}

// B-fragment (32x16) from LDS-staged transposed weight [o][i], padded stride Kp
__device__ __forceinline__ v16h load_b_lds(const h16* w, int Kp, int nt, int kt) {
  int lane = threadIdx.x & 31;
  const h16* p = w + (nt * 16 + (lane & 15)) * Kp + kt + 16 * (lane >> 4);
  v8h lo = *(const v8h*)p;
  v8h hi = *(const v8h*)(p + 8);
  return cat8(lo, hi);
}

// B-fragment straight from global (small kernels only)
__device__ __forceinline__ v16h load_b_g(const h16* w, int K, int nt, int kt) {
  int lane = threadIdx.x & 31;
  const h16* p = w + (size_t)(nt * 16 + (lane & 15)) * K + kt + 16 * (lane >> 4);
  v8h lo = *(const v8h*)p;
  v8h hi = *(const v8h*)(p + 8);
  return cat8(lo, hi);
}

// cooperative copy of one [O][K] f16 weight from global into LDS with stride K+8
__device__ __forceinline__ void stage_w(h16* dst, const h16* __restrict__ src,
                                        int O, int Klog2) {
  int K = 1 << Klog2;
  int Kp = K + 8;
  int total8 = (O << Klog2) >> 3;
  for (int t = threadIdx.x; t < total8; t += blockDim.x) {
    int idx = t << 3;
    int o = idx >> Klog2;
    int i = idx & (K - 1);
    *(v8h*)(dst + o * Kp + i) = *(const v8h*)(src + idx);
  }
}

__device__ __forceinline__ v8f bias_c(const float* b, int nt) {
  float v = b[nt * 16 + (threadIdx.x & 15)];
  v8f c;
  for (int r = 0; r < 8; ++r) c[r] = v;
  return c;
}
__device__ __forceinline__ v8f bias_head(const float* b, int nt) {
  int col = nt * 16 + (threadIdx.x & 15);
  float v = (col < C_) ? b[col] : 0.f;
  v8f c;
  for (int r = 0; r < 8; ++r) c[r] = v;
  return c;
}

__device__ __forceinline__ void store_c_lds(h16* dst, int stride, int colBase,
                                            const v8f& c, int relu) {
  int lane = threadIdx.x & 31;
  int col = colBase + (lane & 15);
  int r0 = 8 * (lane >> 4);
  for (int r = 0; r < 8; ++r) {
    float v = c[r];
    if (relu) v = v > 0.f ? v : 0.f;
    dst[(r0 + r) * stride + col] = (h16)v;
  }
}

__device__ __forceinline__ void store_logits(float* out, int validRows, int nt,
                                             const v8f& c) {
  int lane = threadIdx.x & 31;
  int col = nt * 16 + (lane & 15);
  int r0 = 8 * (lane >> 4);
  if (col < C_) {
    for (int r = 0; r < 8; ++r) {
      int row = r0 + r;
      if (row < validRows) out[(size_t)row * C_ + col] = c[r];
    }
  }
}

__device__ __forceinline__ void load_tile16(h16* dst, const float* src, int validRows) {
  int lane = threadIdx.x & 31;
  for (int it = 0; it < 8; ++it) {
    int idx = it * 32 + lane;
    int row = idx >> 4;
    int col = (idx & 15) * 4;
    float4 v;
    v.x = v.y = v.z = v.w = 0.f;
    if (row < validRows) v = *(const float4*)(src + (size_t)row * 64 + col);
    h16* q = dst + row * ST_X + col;
    q[0] = (h16)v.x; q[1] = (h16)v.y; q[2] = (h16)v.z; q[3] = (h16)v.w;
  }
}

// ---------------- K0: weight convert/transpose f32 -> f16 [s][o][i] -------
__global__ void k_wconv(h16* dst, const float* src, int Scount, int Isrc, int Idst,
                        int Osrc, int Opad) {
  int total = Scount * Opad * Idst;
  int t = blockIdx.x * blockDim.x + threadIdx.x;
  if (t >= total) return;
  int i = t % Idst;
  int o = (t / Idst) % Opad;
  int s = t / (Idst * Opad);
  float v = (o < Osrc) ? src[((size_t)s * Isrc + i) * Osrc + o] : 0.f;
  dst[t] = (h16)v;
}

// ---------------- K1: column means of h and h2 over N ---------------------
// LDS: staging [0,26624) ; wave areas 8x3456 at 26624 ; red floats at 54272
__global__ void k_mean(const float* __restrict__ pts_f, const float* __restrict__ g_img,
                       const h16* __restrict__ Wg, const float* lb, const float* l2b,
                       const float* f1b, const float* f12b, float* acc) {
  extern __shared__ h16 sm[];
  h16* WS = sm;
  int wid = threadIdx.x >> 5;
  h16* wb = sm + 26624 + wid * 3456;       // xp:0, xg:1152, t1:2304
  float* red = (float*)(sm + 54272);       // 128 floats
  int rb = blockIdx.x * 128 + wid * 16;
  int valid = N_ - rb;
  valid = valid > 16 ? 16 : valid;
  if (valid < 0) valid = 0;
  int lane = threadIdx.x & 31;
  int r0 = 8 * (lane >> 4);

  for (int s = 0; s < S_; ++s) {
    __syncthreads();
    if (threadIdx.x < 128) red[threadIdx.x] = 0.f;
    stage_w(WS + 0,     Wg + OFF_LW  + s * 4096, 64, 6);
    stage_w(WS + 4608,  Wg + OFF_LW2 + s * 4096, 64, 6);
    stage_w(WS + 9216,  Wg + OFF_F1  + s * 8192, 64, 7);
    stage_w(WS + 17920, Wg + OFF_F12 + s * 8192, 64, 7);
    __syncthreads();
    if (valid > 0) {
      load_tile16(wb, pts_f + ((size_t)s * N_ + rb) * 64, valid);
      load_tile16(wb + 1152, g_img + ((size_t)s * N_ + rb) * 64, valid);
      h16* t1 = wb + 2304;
      // branch 1: fl = relu(xp@LW+lb); h = [xg|fl]@F1 + f1b
      {
        v16h a0 = load_a_lds(wb, ST_X, 0), a1 = load_a_lds(wb, ST_X, 32);
        for (int nt = 0; nt < 4; ++nt) {
          v8f c = bias_c(lb + s * 64, nt);
          c = wmma_(a0, load_b_lds(WS + 0, 72, nt, 0), c);
          c = wmma_(a1, load_b_lds(WS + 0, 72, nt, 32), c);
          store_c_lds(t1, ST_X, nt * 16, c, 1);
        }
        v16h b0 = load_a_lds(wb + 1152, ST_X, 0), b1 = load_a_lds(wb + 1152, ST_X, 32);
        v16h f0 = load_a_lds(t1, ST_X, 0), f1 = load_a_lds(t1, ST_X, 32);
        for (int nt = 0; nt < 4; ++nt) {
          v8f c = bias_c(f1b + s * 64, nt);
          c = wmma_(b0, load_b_lds(WS + 9216, 136, nt, 0), c);
          c = wmma_(b1, load_b_lds(WS + 9216, 136, nt, 32), c);
          c = wmma_(f0, load_b_lds(WS + 9216, 136, nt, 64), c);
          c = wmma_(f1, load_b_lds(WS + 9216, 136, nt, 96), c);
          float p = 0.f;
          for (int r = 0; r < 8; ++r)
            if (r0 + r < valid) p += c[r];
          atomicAdd(&red[nt * 16 + (lane & 15)], p);
        }
      }
      // branch 2: fl2 = relu(xg@LW2+l2b); h2 = [xp|fl2]@F1_2 + f12b
      {
        v16h a0 = load_a_lds(wb + 1152, ST_X, 0), a1 = load_a_lds(wb + 1152, ST_X, 32);
        for (int nt = 0; nt < 4; ++nt) {
          v8f c = bias_c(l2b + s * 64, nt);
          c = wmma_(a0, load_b_lds(WS + 4608, 72, nt, 0), c);
          c = wmma_(a1, load_b_lds(WS + 4608, 72, nt, 32), c);
          store_c_lds(t1, ST_X, nt * 16, c, 1);
        }
        v16h b0 = load_a_lds(wb, ST_X, 0), b1 = load_a_lds(wb, ST_X, 32);
        v16h f0 = load_a_lds(t1, ST_X, 0), f1 = load_a_lds(t1, ST_X, 32);
        for (int nt = 0; nt < 4; ++nt) {
          v8f c = bias_c(f12b + s * 64, nt);
          c = wmma_(b0, load_b_lds(WS + 17920, 136, nt, 0), c);
          c = wmma_(b1, load_b_lds(WS + 17920, 136, nt, 32), c);
          c = wmma_(f0, load_b_lds(WS + 17920, 136, nt, 64), c);
          c = wmma_(f1, load_b_lds(WS + 17920, 136, nt, 96), c);
          float p = 0.f;
          for (int r = 0; r < 8; ++r)
            if (r0 + r < valid) p += c[r];
          atomicAdd(&red[64 + nt * 16 + (lane & 15)], p);
        }
      }
    }
    __syncthreads();
    if (threadIdx.x < 64)
      atomicAdd(&acc[s * 64 + threadIdx.x], red[threadIdx.x]);
    else if (threadIdx.x < 128)
      atomicAdd(&acc[256 + s * 64 + (threadIdx.x - 64)], red[threadIdx.x]);
  }
}

// ---------------- K1b: c2[s] = (mean)@fcs2_W[64:] + fcs2_b ----------------
__global__ void k_c2(const float* f2W, const float* f2b, const float* f22W,
                     const float* f22b, const float* acc, float* c2, float* c22) {
  int tid = blockIdx.x * blockDim.x + threadIdx.x;
  if (tid >= S_ * 64) return;
  int s = tid >> 6, o = tid & 63;
  float v = f2b[s * 64 + o];
  float v2 = f22b[s * 64 + o];
  const float inv = 1.f / (float)N_;
  for (int i = 0; i < 64; ++i) {
    float m = acc[s * 64 + i] * inv;
    float m2 = acc[256 + s * 64 + i] * inv;
    v += m * f2W[((size_t)s * 128 + 64 + i) * 64 + o];
    v2 += m2 * f22W[((size_t)s * 128 + 64 + i) * 64 + o];
  }
  c2[s * 64 + o] = v;
  c22[s * 64 + o] = v2;
}

// ---------------- fused fuse-gate pipeline (16 rows, one wave) ------------
__device__ __forceinline__ void do_fuse(h16* wb, int xAoff, int xBoff, h16* cat,
    int catCol, const h16* wLW, const h16* wF1, const h16* wF2a,
    const float* lbv, const float* f1bv, const float* c2v) {
  h16* t1 = wb + WV_T1;
  // fl = relu(xA @ LW + lb) -> t1
  {
    v16h a0 = load_a_lds(wb + xAoff, ST_X, 0), a1 = load_a_lds(wb + xAoff, ST_X, 32);
    for (int nt = 0; nt < 4; ++nt) {
      v8f c = bias_c(lbv, nt);
      c = wmma_(a0, load_b_lds(wLW, 72, nt, 0), c);
      c = wmma_(a1, load_b_lds(wLW, 72, nt, 32), c);
      store_c_lds(t1, ST_X, nt * 16, c, 1);
    }
  }
  // h = [xB | fl] @ F1 + f1b (regs + t1 for A re-read)
  v8f hc[4];
  {
    v16h b0 = load_a_lds(wb + xBoff, ST_X, 0), b1 = load_a_lds(wb + xBoff, ST_X, 32);
    v16h f0 = load_a_lds(t1, ST_X, 0), f1 = load_a_lds(t1, ST_X, 32);
    for (int nt = 0; nt < 4; ++nt) {
      v8f c = bias_c(f1bv, nt);
      c = wmma_(b0, load_b_lds(wF1, 136, nt, 0), c);
      c = wmma_(b1, load_b_lds(wF1, 136, nt, 32), c);
      c = wmma_(f0, load_b_lds(wF1, 136, nt, 64), c);
      c = wmma_(f1, load_b_lds(wF1, 136, nt, 96), c);
      hc[nt] = c;
    }
    for (int nt = 0; nt < 4; ++nt) store_c_lds(t1, ST_X, nt * 16, hc[nt], 0);
  }
  // w = sigmoid(h @ F2a + c2) ; fuse = relu(h*w)+xB -> cat
  {
    v16h h0 = load_a_lds(t1, ST_X, 0), h1 = load_a_lds(t1, ST_X, 32);
    int lane = threadIdx.x & 31;
    int r0 = 8 * (lane >> 4);
    for (int nt = 0; nt < 4; ++nt) {
      v8f c = bias_c(c2v, nt);
      c = wmma_(h0, load_b_lds(wF2a, 72, nt, 0), c);
      c = wmma_(h1, load_b_lds(wF2a, 72, nt, 32), c);
      int col = nt * 16 + (lane & 15);
      for (int r = 0; r < 8; ++r) {
        float g = 1.f / (1.f + __expf(-c[r]));
        float v = hc[nt][r] * g;
        v = v > 0.f ? v : 0.f;
        v += (float)wb[xBoff + (r0 + r) * ST_X + col];
        cat[(r0 + r) * ST_CAT + catCol + col] = (h16)v;
      }
    }
  }
}

// head over one 64-col slice of cat (hidden buffer aliases dead xp/xg)
__device__ __forceinline__ void do_head(h16* wb, h16* cat, int catCol,
    const h16* wH1, const h16* wH2, const float* hb1, const float* hb2,
    float* outHead, int valid) {
  h16* tb = wb;  // 16 x 136 over dead xp/xg
  v16h a0 = load_a_lds(cat, ST_CAT, catCol), a1 = load_a_lds(cat, ST_CAT, catCol + 32);
  for (int nt = 0; nt < 8; ++nt) {
    v8f c = bias_c(hb1, nt);
    c = wmma_(a0, load_b_lds(wH1, 72, nt, 0), c);
    c = wmma_(a1, load_b_lds(wH1, 72, nt, 32), c);
    store_c_lds(tb, ST_TB, nt * 16, c, 1);
  }
  v16h t0 = load_a_lds(tb, ST_TB, 0), t1 = load_a_lds(tb, ST_TB, 32);
  v16h t2 = load_a_lds(tb, ST_TB, 64), t3 = load_a_lds(tb, ST_TB, 96);
  for (int nt = 0; nt < 2; ++nt) {
    v8f c = bias_head(hb2, nt);
    c = wmma_(t0, load_b_lds(wH2, 136, nt, 0), c);
    c = wmma_(t1, load_b_lds(wH2, 136, nt, 32), c);
    c = wmma_(t2, load_b_lds(wH2, 136, nt, 64), c);
    c = wmma_(t3, load_b_lds(wH2, 136, nt, 96), c);
    store_logits(outHead, valid, nt, c);
  }
}

__device__ __forceinline__ void do_cls(h16* wb, h16* cat, const h16* w1,
                                       const h16* w2, const float* b1,
                                       const float* b2, float* out, int valid) {
  h16* tb = wb;
  v16h a[8];
  for (int k = 0; k < 8; ++k) a[k] = load_a_lds(cat, ST_CAT, k * 32);
  for (int nt = 0; nt < 8; ++nt) {
    v8f c = bias_c(b1, nt);
    for (int k = 0; k < 8; ++k) c = wmma_(a[k], load_b_lds(w1, 264, nt, k * 32), c);
    store_c_lds(tb, ST_TB, nt * 16, c, 1);
  }
  v16h t0 = load_a_lds(tb, ST_TB, 0), t1 = load_a_lds(tb, ST_TB, 32);
  v16h t2 = load_a_lds(tb, ST_TB, 64), t3 = load_a_lds(tb, ST_TB, 96);
  for (int nt = 0; nt < 2; ++nt) {
    v8f c = bias_head(b2, nt);
    c = wmma_(t0, load_b_lds(w2, 136, nt, 0), c);
    c = wmma_(t1, load_b_lds(w2, 136, nt, 32), c);
    c = wmma_(t2, load_b_lds(w2, 136, nt, 64), c);
    c = wmma_(t3, load_b_lds(w2, 136, nt, 96), c);
    store_logits(out, valid, nt, c);
  }
}

// ---------------- K2: fully fused main pipeline ---------------------------
__global__ void k_main(const float* __restrict__ pts_f, const float* __restrict__ g_img,
                       const h16* __restrict__ Wg, const float* lb, const float* l2b,
                       const float* f1b, const float* f12b, const float* hfb1,
                       const float* hfb2, const float* hf2b1, const float* hf2b2,
                       const float* clsb1, const float* clsb2, const float* cls2b1,
                       const float* cls2b2, const float* c2, const float* c22,
                       float* out_img, float* out_pts, float* out_fg, float* out_fp) {
  extern __shared__ h16 sm[];
  h16* WS = sm;                               // weight staging
  int wid = threadIdx.x >> 5;
  h16* wb = sm + WS_HALVES + wid * WV_HALVES; // per-wave working area
  int rb = blockIdx.x * 128 + wid * 16;
  int valid = N_ - rb;
  valid = valid > 16 ? 16 : valid;
  if (valid < 0) valid = 0;
  h16* cat1 = wb + WV_CAT1;
  h16* cat2 = wb + WV_CAT2;

  for (int s = 0; s < S_; ++s) {
    __syncthreads();
    stage_w(WS + LS_LW,    Wg + OFF_LW    + s * 4096, 64, 6);
    stage_w(WS + LS_LW2,   Wg + OFF_LW2   + s * 4096, 64, 6);
    stage_w(WS + LS_F1,    Wg + OFF_F1    + s * 8192, 64, 7);
    stage_w(WS + LS_F12,   Wg + OFF_F12   + s * 8192, 64, 7);
    stage_w(WS + LS_F2A,   Wg + OFF_F2A   + s * 4096, 64, 6);
    stage_w(WS + LS_F2A2,  Wg + OFF_F2A2  + s * 4096, 64, 6);
    stage_w(WS + LS_HFW1,  Wg + OFF_HFW1  + s * 8192, 128, 6);
    stage_w(WS + LS_HF2W1, Wg + OFF_HF2W1 + s * 8192, 128, 6);
    stage_w(WS + LS_HFW2,  Wg + OFF_HFW2  + s * 4096, 32, 7);
    stage_w(WS + LS_HF2W2, Wg + OFF_HF2W2 + s * 4096, 32, 7);
    __syncthreads();
    if (valid > 0) {
      load_tile16(wb + WV_XP, pts_f + ((size_t)s * N_ + rb) * 64, valid);
      load_tile16(wb + WV_XG, g_img + ((size_t)s * N_ + rb) * 64, valid);
      // both fuse stages first (xp/xg live), then heads (hidden aliases xp/xg)
      do_fuse(wb, WV_XP, WV_XG, cat1, s * 64, WS + LS_LW, WS + LS_F1, WS + LS_F2A,
              lb + s * 64, f1b + s * 64, c2 + s * 64);
      do_fuse(wb, WV_XG, WV_XP, cat2, s * 64, WS + LS_LW2, WS + LS_F12, WS + LS_F2A2,
              l2b + s * 64, f12b + s * 64, c22 + s * 64);
      do_head(wb, cat1, s * 64, WS + LS_HFW1, WS + LS_HFW2, hfb1 + s * 128,
              hfb2 + s * C_, out_fg + ((size_t)s * N_ + rb) * C_, valid);
      do_head(wb, cat2, s * 64, WS + LS_HF2W1, WS + LS_HF2W2, hf2b1 + s * 128,
              hf2b2 + s * C_, out_fp + ((size_t)s * N_ + rb) * C_, valid);
    }
  }
  // cls head 1
  __syncthreads();
  stage_w(WS, Wg + OFF_CLS1, 128, 8);
  stage_w(WS + LS_CLSW2, Wg + OFF_CLSW2, 32, 7);
  __syncthreads();
  if (valid > 0)
    do_cls(wb, cat1, WS, WS + LS_CLSW2, clsb1, clsb2, out_img + (size_t)rb * C_, valid);
  // cls head 2
  __syncthreads();
  stage_w(WS, Wg + OFF_CLS21, 128, 8);
  stage_w(WS + LS_CLSW2, Wg + OFF_CLS22, 32, 7);
  __syncthreads();
  if (valid > 0)
    do_cls(wb, cat2, WS, WS + LS_CLSW2, cls2b1, cls2b2, out_pts + (size_t)rb * C_, valid);
}

// ---------------- K3: head3d with double gather ---------------------------
__global__ void k_head3d(const float* __restrict__ pts_feat, const int* coors_inv,
                         const int* p2idx, const h16* __restrict__ W, const float* b1,
                         const float* b2, float* out) {
  extern __shared__ h16 sm[];
  int wid = threadIdx.x >> 5;
  h16* xb = sm + wid * 3328;
  h16* tb = xb + 1152;
  int s = blockIdx.y;
  int rb = blockIdx.x * 128 + wid * 16;
  int lane = threadIdx.x & 31;
  for (int it = 0; it < 8; ++it) {
    int idx = it * 32 + lane;
    int row = idx >> 4;
    int col = (idx & 15) * 4;
    int j = rb + row;
    int r = coors_inv[(size_t)s * N_ + p2idx[j]];
    float4 v = *(const float4*)(pts_feat + ((size_t)s * NV_ + r) * 64 + col);
    h16* q = xb + row * ST_X + col;
    q[0] = (h16)v.x; q[1] = (h16)v.y; q[2] = (h16)v.z; q[3] = (h16)v.w;
  }
  const h16* w1 = W + OFF_H3W1 + s * 8192;
  v16h a0 = load_a_lds(xb, ST_X, 0), a1 = load_a_lds(xb, ST_X, 32);
  for (int nt = 0; nt < 8; ++nt) {
    v8f c = bias_c(b1 + s * 128, nt);
    c = wmma_(a0, load_b_g(w1, 64, nt, 0), c);
    c = wmma_(a1, load_b_g(w1, 64, nt, 32), c);
    store_c_lds(tb, ST_TB, nt * 16, c, 1);
  }
  const h16* w2 = W + OFF_H3W2 + s * 4096;
  v16h t0 = load_a_lds(tb, ST_TB, 0), t1 = load_a_lds(tb, ST_TB, 32);
  v16h t2 = load_a_lds(tb, ST_TB, 64), t3 = load_a_lds(tb, ST_TB, 96);
  for (int nt = 0; nt < 2; ++nt) {
    v8f c = bias_head(b2 + s * C_, nt);
    c = wmma_(t0, load_b_g(w2, 128, nt, 0), c);
    c = wmma_(t1, load_b_g(w2, 128, nt, 32), c);
    c = wmma_(t2, load_b_g(w2, 128, nt, 64), c);
    c = wmma_(t3, load_b_g(w2, 128, nt, 96), c);
    store_logits(out + ((size_t)s * M_ + rb) * C_, 16, nt, c);
  }
}

// ---------------- K4: g_loss ----------------------------------------------
__global__ void k_gloss(const float* __restrict__ g_img, const float* __restrict__ img_feat,
                        const int* p2idx, float* outScalar) {
  __shared__ float red;
  if (threadIdx.x == 0) red = 0.f;
  __syncthreads();
  size_t i = (size_t)blockIdx.x * blockDim.x + threadIdx.x;
  const size_t total = (size_t)S_ * M_ * 64;
  float v = 0.f;
  if (i < total) {
    int c = (int)(i & 63);
    size_t t = i >> 6;
    int j = (int)(t % M_);
    int s = (int)(t / M_);
    float a = g_img[((size_t)s * N_ + p2idx[j]) * 64 + c];
    float b = img_feat[((size_t)s * M_ + j) * 64 + c];
    float d = a - b;
    v = d * d;
  }
  for (int off = 16; off; off >>= 1) v += __shfl_down(v, off, 32);
  if ((threadIdx.x & 31) == 0) atomicAdd(&red, v);
  __syncthreads();
  if (threadIdx.x == 0)
    atomicAdd(outScalar, red * (1.f / ((float)S_ * (float)M_ * 64.f)));
}

// ---------------- host launcher -------------------------------------------
extern "C" void kernel_launch(void* const* d_in, const int* in_sizes, int n_in,
                              void* d_out, int out_size, void* d_ws, size_t ws_size,
                              hipStream_t stream) {
  const float* pts_feat_f = (const float*)d_in[0];
  const float* g_img_feat = (const float*)d_in[1];
  const float* pts_feat   = (const float*)d_in[2];
  const float* img_feat   = (const float*)d_in[3];
  const int*   coors_inv  = (const int*)d_in[4];
  const int*   p2img_idx  = (const int*)d_in[5];
  const float* h3b1 = (const float*)d_in[7];
  const float* h3b2 = (const float*)d_in[9];
  const float* hfb1 = (const float*)d_in[11];
  const float* hfb2 = (const float*)d_in[13];
  const float* hf2b1 = (const float*)d_in[15];
  const float* hf2b2 = (const float*)d_in[17];
  const float* lb   = (const float*)d_in[19];
  const float* l2b  = (const float*)d_in[21];
  const float* f1b  = (const float*)d_in[23];
  const float* f2W  = (const float*)d_in[24];
  const float* f2b  = (const float*)d_in[25];
  const float* f12b = (const float*)d_in[27];
  const float* f22W = (const float*)d_in[28];
  const float* f22b = (const float*)d_in[29];
  const float* clsb1 = (const float*)d_in[31];
  const float* clsb2 = (const float*)d_in[33];
  const float* cls2b1 = (const float*)d_in[35];
  const float* cls2b2 = (const float*)d_in[37];

  h16* W = (h16*)d_ws;
  float* acc = (float*)((char*)d_ws + (size_t)WTOT * sizeof(h16));
  float* c2 = acc + 512;
  float* c22 = c2 + 256;

  float* out_img = (float*)d_out;
  float* out_pts = out_img + (size_t)N_ * C_;
  float* out_fg  = out_pts + (size_t)N_ * C_;
  float* out_fp  = out_fg + (size_t)S_ * N_ * C_;
  float* out_pred = out_fp + (size_t)S_ * N_ * C_;
  float* out_gloss = out_pred + (size_t)S_ * M_ * C_;

  hipMemsetAsync(acc, 0, 512 * sizeof(float), stream);
  hipMemsetAsync(out_gloss, 0, sizeof(float), stream);

  auto conv = [&](int off, int inIdx, int Sc, int Isrc, int Idst, int Osrc, int Opad) {
    int total = Sc * Opad * Idst;
    k_wconv<<<(total + 255) / 256, 256, 0, stream>>>(
        W + off, (const float*)d_in[inIdx], Sc, Isrc, Idst, Osrc, Opad);
  };
  conv(OFF_LW,   18, 4,  64,  64,  64,  64);
  conv(OFF_LW2,  20, 4,  64,  64,  64,  64);
  conv(OFF_F1,   22, 4, 128, 128,  64,  64);
  conv(OFF_F12,  26, 4, 128, 128,  64,  64);
  conv(OFF_F2A,  24, 4, 128,  64,  64,  64);
  conv(OFF_F2A2, 28, 4, 128,  64,  64,  64);
  conv(OFF_H3W1,  6, 4,  64,  64, 128, 128);
  conv(OFF_H3W2,  8, 4, 128, 128,  20,  32);
  conv(OFF_HFW1, 10, 4,  64,  64, 128, 128);
  conv(OFF_HFW2, 12, 4, 128, 128,  20,  32);
  conv(OFF_HF2W1,14, 4,  64,  64, 128, 128);
  conv(OFF_HF2W2,16, 4, 128, 128,  20,  32);
  conv(OFF_CLS1, 30, 1, 256, 256, 128, 128);
  conv(OFF_CLSW2,32, 1, 128, 128,  20,  32);
  conv(OFF_CLS21,34, 1, 256, 256, 128, 128);
  conv(OFF_CLS22,36, 1, 128, 128,  20,  32);

  int nb = (N_ + 127) / 128;
  k_mean<<<nb, 256, 54272 * 2 + 512, stream>>>(pts_feat_f, g_img_feat, W, lb, l2b,
                                               f1b, f12b, acc);
  k_c2<<<1, 256, 0, stream>>>(f2W, f2b, f22W, f22b, acc, c2, c22);
  k_main<<<nb, 256, (WS_HALVES + 8 * WV_HALVES) * 2, stream>>>(
      pts_feat_f, g_img_feat, W, lb, l2b, f1b, f12b, hfb1, hfb2, hf2b1, hf2b2,
      clsb1, clsb2, cls2b1, cls2b2, c2, c22, out_img, out_pts, out_fg, out_fp);
  k_head3d<<<dim3(M_ / 128, S_), 256, 8 * 3328 * 2, stream>>>(
      pts_feat, coors_inv, p2img_idx, W, h3b1, h3b2, out_pred);
  size_t gl = (size_t)S_ * M_ * 64;
  k_gloss<<<(int)((gl + 255) / 256), 256, 0, stream>>>(g_img_feat, img_feat,
                                                       p2img_idx, out_gloss);
  (void)in_sizes; (void)n_in; (void)out_size; (void)ws_size;
}
